// EncoderLayer_1589137899705
// MI455X (gfx1250) — matrix operand order repaired and checked by reading, not compiled, not run on hardware
//
#include <hip/hip_runtime.h>

// ---------------------------------------------------------------------------
// CDNA5 / gfx1250 sparse 3D U-Net.
// C=16 -> C=16 convs = implicit GEMM on V_WMMA_F32_16X16X32_F16:
//   M = 16 voxels (tile of consecutive linear voxel indices, channels-last)
//   N = 16 output channels, K = 32 = 2 conv taps x 16 input channels.
// Weights staged global -> LDS by the Tensor Data Mover (TENSORcnt); B
// fragments are served from LDS (ds_load_b128). Halo handling is branchless
// (clamp + cndmask) so EXEC stays all-ones around the WMMAs.
// ---------------------------------------------------------------------------

typedef __attribute__((ext_vector_type(16))) _Float16 v16h;
typedef __attribute__((ext_vector_type(8)))  _Float16 v8h;
typedef __attribute__((ext_vector_type(8)))  float    v8f;
typedef __attribute__((ext_vector_type(4)))  unsigned int u32x4;
typedef __attribute__((ext_vector_type(8)))  int          i32x8;
typedef __attribute__((ext_vector_type(4)))  int          i32x4;

#if __has_builtin(__builtin_amdgcn_tensor_load_to_lds) && __has_builtin(__builtin_amdgcn_s_wait_tensorcnt)
#define USE_TDM 1
#else
#define USE_TDM 0
#endif

// Stage a weight blob (nbytes, multiple of 16) into LDS at offset 0.
// TDM path: 1-D tensor descriptor, data_size=8B, tile==tensor.
__device__ inline void stage_weights(const _Float16* __restrict__ wq,
                                     _Float16* wlds, int nbytes) {
#if USE_TDM
  if (threadIdx.x < 32) {   // one wave issues the DMA (TDM is wave-level)
    unsigned long long ga = (unsigned long long)wq;
    unsigned n8 = (unsigned)nbytes >> 3;              // 8-byte elements
    u32x4 g0 = {};
    g0[0] = 1u;                                       // count=1 valid descriptor
    g0[1] = 0u;                                       // lds_addr = 0
    g0[2] = (unsigned)(ga & 0xffffffffu);             // global_addr[31:0]
    g0[3] = (unsigned)((ga >> 32) & 0x01ffffffu) | (2u << 30); // addr[56:32] | type=2
    i32x8 g1 = {};
    g1[0] = (int)(3u << 16);                          // data_size=8B; no multicast/pad
    g1[1] = (int)((n8 & 0xffffu) << 16);              // tensor_dim0 lo
    g1[2] = (int)((n8 >> 16) | (1u << 16));           // tensor_dim0 hi | tensor_dim1=1
    g1[3] = (int)(n8 << 16);                          // tile_dim0 = n8
    g1[4] = 0;                                        // tile_dim1/2 unused
    g1[5] = (int)n8;                                  // tensor_dim0_stride lo
    g1[6] = 0; g1[7] = 0;
    i32x4 z4 = {};
#if defined(__clang_major__) && __clang_major__ >= 23
    i32x8 z8 = {};
    __builtin_amdgcn_tensor_load_to_lds(g0, g1, z4, z4, z8, 0);
#else
    __builtin_amdgcn_tensor_load_to_lds(g0, g1, z4, z4, 0);
#endif
    __builtin_amdgcn_s_wait_tensorcnt((short)0);
  }
#else
  const int n16 = nbytes >> 4;
  for (int i = threadIdx.x; i < n16; i += blockDim.x)
    ((v8h*)wlds)[i] = ((const v8h*)wq)[i];
#endif
  __syncthreads();
  // Escape the LDS pointer: the TDM write is invisible to alias analysis, so
  // without this the compiler folds the (apparently never-written) LDS reads.
  asm volatile("" : : "v"(wlds) : "memory");
}

// A-fragment half-row load with 32-bit byte offset (saddr-form addressing).
__device__ inline v8h load_a8(const _Float16* __restrict__ in, unsigned voxel, int hi) {
  return *(const v8h*)((const char*)in + ((size_t)((voxel << 5) + (unsigned)(hi << 4))));
}

// ------------------------- input grid build --------------------------------
__global__ void k_scatter(const float* __restrict__ data, const int* __restrict__ clust,
                          float* __restrict__ cnt, int n) {
  int i = blockIdx.x * blockDim.x + threadIdx.x;
  if (i >= n) return;
  int x = (int)data[i * 5 + 0];
  int y = (int)data[i * 5 + 1];
  int z = (int)data[i * 5 + 2];
  int b = clust[i];
  int lin = ((((b << 6) | x) << 6) | y) << 6 | z;
  atomicAdd(&cnt[lin], 1.0f);
}

__global__ void k_mask_from_cnt(const float* __restrict__ cnt, unsigned char* __restrict__ mask, int nv) {
  int v = blockIdx.x * blockDim.x + threadIdx.x;
  if (v >= nv) return;
  mask[v] = (cnt[v] > 0.0f) ? 1 : 0;
}

// prepare: SubmanifoldConvolution 1 -> 16, k=3, masked (tiny work, scalar).
__global__ void k_prep_conv(const float* __restrict__ cnt, const unsigned char* __restrict__ mask,
                            const float* __restrict__ w, _Float16* __restrict__ out, int nv) {
  int v = blockIdx.x * blockDim.x + threadIdx.x;
  if (v >= nv) return;
  float acc[16];
#pragma unroll
  for (int c = 0; c < 16; ++c) acc[c] = 0.0f;
  if (mask[v]) {
    const int Z = v & 63, Y = (v >> 6) & 63, X = (v >> 12) & 63, Bb = v >> 18;
    for (int t = 0; t < 27; ++t) {
      const int dx = t / 9 - 1, dy = (t / 3) % 3 - 1, dz = t % 3 - 1;
      const int nx = X + dx, ny = Y + dy, nz = Z + dz;
      if ((unsigned)nx < 64u && (unsigned)ny < 64u && (unsigned)nz < 64u) {
        const float cv = cnt[((((Bb << 6) | nx) << 6) | ny) << 6 | nz];
        if (cv != 0.0f) {
#pragma unroll
          for (int c = 0; c < 16; ++c) acc[c] += cv * w[t * 16 + c];
        }
      }
    }
  }
#pragma unroll
  for (int c = 0; c < 16; ++c) out[(size_t)v * 16 + c] = (_Float16)acc[c];
}

// ------------------------- weight repack to f16 ----------------------------
// src (f32): [tap][ci][co]. dst (f16): [pair][slot][co][ci].
// mode 0: 2 taps per pair (K=32), tail zero-padded. mode 1: 1 tap/pair, slot1=0.
__global__ void k_wprep(const float* __restrict__ src, _Float16* __restrict__ dst,
                        int ntaps, int mode) {
  int i = blockIdx.x * blockDim.x + threadIdx.x;
  int npairs = mode ? ntaps : (ntaps + 1) / 2;
  if (i >= npairs * 512) return;
  int ci = i & 15, co = (i >> 4) & 15, slot = (i >> 8) & 1, pair = i >> 9;
  int tap = mode ? pair : pair * 2 + slot;
  float v = 0.0f;
  if ((mode == 0 && tap < ntaps) || (mode == 1 && slot == 0))
    v = src[tap * 256 + ci * 16 + co];
  dst[i] = (_Float16)v;
}

// ------------------------- BN over active sites ----------------------------
__global__ void k_bn_stats(const _Float16* __restrict__ x, const unsigned char* __restrict__ mask,
                           float* __restrict__ stats, int nv) {
  __shared__ float ls[256], lq[256], lcnt[16];
  const int tid = blockIdx.x * blockDim.x + threadIdx.x;
  const int c = threadIdx.x & 15;
  const int g = threadIdx.x >> 4;
  const int stride = gridDim.x * blockDim.x;      // multiple of 16 -> fixed channel
  const int total = nv * 16;
  float s = 0.0f, q = 0.0f, n = 0.0f;
  for (int e = tid; e < total; e += stride) {
    float xv = (float)x[e];
    s += xv; q += xv * xv;
    if (c == 0) n += mask[e >> 4] ? 1.0f : 0.0f;
  }
  ls[threadIdx.x] = s; lq[threadIdx.x] = q;
  if (c == 0) lcnt[g] = n;
  __syncthreads();
  if (g == 0) {
    float ss = 0.0f, qq = 0.0f;
    for (int k = 0; k < 16; ++k) { ss += ls[k * 16 + c]; qq += lq[k * 16 + c]; }
    atomicAdd(&stats[c], ss);
    atomicAdd(&stats[16 + c], qq);
    if (c == 0) {
      float nn = 0.0f;
      for (int k = 0; k < 16; ++k) nn += lcnt[k];
      atomicAdd(&stats[32], nn);
    }
  }
}

__global__ void k_bn_finalize(const float* __restrict__ stats, const float* __restrict__ gamma,
                              const float* __restrict__ beta, float* __restrict__ ss) {
  int c = threadIdx.x;
  if (c >= 16) return;
  float n = fmaxf(stats[32], 1.0f);
  float mean = stats[c] / n;
  float var = stats[16 + c] / n - mean * mean;
  float sc = gamma[c] * rsqrtf(var + 1e-4f);
  ss[c] = sc;
  ss[16 + c] = beta[c] - mean * sc;
}

__global__ void k_bn_apply(_Float16* __restrict__ x, const unsigned char* __restrict__ mask,
                           const float* __restrict__ ss, int nv) {
  int e = blockIdx.x * blockDim.x + threadIdx.x;
  if (e >= nv * 16) return;
  int c = e & 15, v = e >> 4;
  float xv = (float)x[e] * ss[c] + ss[16 + c];
  xv = xv > 0.0f ? xv : 0.0f;                    // LEAK = 0
  x[e] = (_Float16)(mask[v] ? xv : 0.0f);
}

// ------------------------- mask ops ----------------------------------------
__global__ void k_mask_down(const unsigned char* __restrict__ mi, unsigned char* __restrict__ mo,
                            int sho, int nvo) {
  int v = blockIdx.x * blockDim.x + threadIdx.x;
  if (v >= nvo) return;
  const int DH = 1 << sho, shi = sho + 1;
  const int Z = v & (DH - 1), Y = (v >> sho) & (DH - 1), X = (v >> (2 * sho)) & (DH - 1);
  const int Bb = v >> (3 * sho);
  unsigned char m = 0;
  for (int t = 0; t < 8; ++t) {
    const int dx = t >> 2, dy = (t >> 1) & 1, dz = t & 1;
    const int nv = ((((Bb << shi) | (2 * X + dx)) << shi) | (2 * Y + dy)) << shi | (2 * Z + dz);
    m |= mi[nv];
  }
  mo[v] = m ? 1 : 0;
}

__global__ void k_sparsify(_Float16* __restrict__ x, unsigned char* __restrict__ mask, int nv) {
  int v = blockIdx.x * blockDim.x + threadIdx.x;
  if (v >= nv) return;
  unsigned char m = (mask[v] && ((float)x[(size_t)v * 16] > 0.0f)) ? 1 : 0;
  mask[v] = m;
  if (!m) {
#pragma unroll
    for (int c = 0; c < 16; ++c) x[(size_t)v * 16 + c] = (_Float16)0.0f;
  }
}

// ------------------------- WMMA convolutions -------------------------------
// k=3 pad=1 submanifold conv. 27 taps -> 14 K=32 WMMAs (tap 27 = zeros).
__global__ __launch_bounds__(128) void k_conv3_wmma(
    const _Float16* __restrict__ in, _Float16* __restrict__ out,
    const unsigned char* __restrict__ mask, const _Float16* __restrict__ wq, int sh) {
  __shared__ __align__(64) _Float16 wlds[14 * 512];
  stage_weights(wq, wlds, 14 * 1024);
  const int lane = threadIdx.x & 31;
  const int tile = blockIdx.x * (blockDim.x >> 5) + (threadIdx.x >> 5);
  const int base = tile << 4;
  const int DD = 1 << sh;
  const int ln = lane & 15, hi = lane >> 4;
  const int vox = base + ln;
  const int Z = vox & (DD - 1), Y = (vox >> sh) & (DD - 1), X = (vox >> (2 * sh)) & (DD - 1);
  const int Bb = vox >> (3 * sh);
  __builtin_prefetch(in + (size_t)(base + 16) * 16, 0, 1);   // next tile
  const v8h zero = {};
  v8f acc = {};
  for (int p = 0; p < 14; ++p) {
    v16h a;
#pragma unroll
    for (int s = 0; s < 2; ++s) {
      const int t = 2 * p + s;
      const int dx = t / 9 - 1, dy = (t / 3) % 3 - 1, dz = t % 3 - 1;
      const int nx = X + dx, ny = Y + dy, nz = Z + dz;
      const bool ok = (t < 27) &
                      ((unsigned)nx < (unsigned)DD) & ((unsigned)ny < (unsigned)DD) &
                      ((unsigned)nz < (unsigned)DD);
      const int cx = min(max(nx, 0), DD - 1);    // clamped -> always legal address
      const int cy = min(max(ny, 0), DD - 1);
      const int cz = min(max(nz, 0), DD - 1);
      const unsigned nvx = (unsigned)(((((Bb << sh) | cx) << sh) | cy) << sh | cz);
      v8h h = load_a8(in, nvx, hi);
      h = ok ? h : zero;                          // branchless halo: EXEC untouched
#pragma unroll
      for (int q = 0; q < 8; ++q) a[s * 8 + q] = h[q];
    }
    const v8h b0 = *(const v8h*)(wlds + (size_t)((p * 2) * 16 + ln) * 16 + hi * 8);
    const v8h b1 = *(const v8h*)(wlds + (size_t)((p * 2 + 1) * 16 + ln) * 16 + hi * 8);
    const v16h b = __builtin_shufflevector(b0, b1, 0, 1, 2, 3, 4, 5, 6, 7, 8, 9, 10, 11, 12, 13, 14, 15);
    acc = __builtin_amdgcn_wmma_f32_16x16x32_f16(false, a, false, b, (short)0, acc, false, false);
  }
#pragma unroll
  for (int r = 0; r < 8; ++r) {
    const int ov = base + hi * 8 + r;
    const float mk = mask[ov] ? 1.0f : 0.0f;
    out[(size_t)ov * 16 + ln] = (_Float16)(acc[r] * mk);
  }
}

// k=4 pad (1,2) submanifold conv: offsets -1..2. 64 taps -> 32 WMMAs.
__global__ __launch_bounds__(128) void k_conv4_wmma(
    const _Float16* __restrict__ in, _Float16* __restrict__ out,
    const unsigned char* __restrict__ mask, const _Float16* __restrict__ wq, int sh) {
  __shared__ __align__(64) _Float16 wlds[32 * 512];
  stage_weights(wq, wlds, 32 * 1024);
  const int lane = threadIdx.x & 31;
  const int tile = blockIdx.x * (blockDim.x >> 5) + (threadIdx.x >> 5);
  const int base = tile << 4;
  const int DD = 1 << sh;
  const int ln = lane & 15, hi = lane >> 4;
  const int vox = base + ln;
  const int Z = vox & (DD - 1), Y = (vox >> sh) & (DD - 1), X = (vox >> (2 * sh)) & (DD - 1);
  const int Bb = vox >> (3 * sh);
  const v8h zero = {};
  v8f acc = {};
  for (int p = 0; p < 32; ++p) {
    v16h a;
#pragma unroll
    for (int s = 0; s < 2; ++s) {
      const int t = 2 * p + s;
      const int dx = (t >> 4) - 1, dy = ((t >> 2) & 3) - 1, dz = (t & 3) - 1;
      const int nx = X + dx, ny = Y + dy, nz = Z + dz;
      const bool ok = ((unsigned)nx < (unsigned)DD) & ((unsigned)ny < (unsigned)DD) &
                      ((unsigned)nz < (unsigned)DD);
      const int cx = min(max(nx, 0), DD - 1);
      const int cy = min(max(ny, 0), DD - 1);
      const int cz = min(max(nz, 0), DD - 1);
      const unsigned nvx = (unsigned)(((((Bb << sh) | cx) << sh) | cy) << sh | cz);
      v8h h = load_a8(in, nvx, hi);
      h = ok ? h : zero;
#pragma unroll
      for (int q = 0; q < 8; ++q) a[s * 8 + q] = h[q];
    }
    const v8h b0 = *(const v8h*)(wlds + (size_t)((p * 2) * 16 + ln) * 16 + hi * 8);
    const v8h b1 = *(const v8h*)(wlds + (size_t)((p * 2 + 1) * 16 + ln) * 16 + hi * 8);
    const v16h b = __builtin_shufflevector(b0, b1, 0, 1, 2, 3, 4, 5, 6, 7, 8, 9, 10, 11, 12, 13, 14, 15);
    acc = __builtin_amdgcn_wmma_f32_16x16x32_f16(false, a, false, b, (short)0, acc, false, false);
  }
#pragma unroll
  for (int r = 0; r < 8; ++r) {
    const int ov = base + hi * 8 + r;
    const float mk = mask[ov] ? 1.0f : 0.0f;
    out[(size_t)ov * 16 + ln] = (_Float16)(acc[r] * mk);
  }
}

// Strided conv k=2 s=2 VALID. Tile over OUTPUT voxels; 8 taps -> 4 WMMAs.
__global__ __launch_bounds__(128) void k_down_wmma(
    const _Float16* __restrict__ in, _Float16* __restrict__ out,
    const unsigned char* __restrict__ maskOut, const _Float16* __restrict__ wq, int sh) {
  __shared__ __align__(64) _Float16 wlds[4 * 512];
  stage_weights(wq, wlds, 4 * 1024);
  const int lane = threadIdx.x & 31;
  const int tile = blockIdx.x * (blockDim.x >> 5) + (threadIdx.x >> 5);
  const int base = tile << 4;
  const int DH = 1 << sh, shi = sh + 1;
  const int ln = lane & 15, hi = lane >> 4;
  const int vox = base + ln;
  const int Z = vox & (DH - 1), Y = (vox >> sh) & (DH - 1), X = (vox >> (2 * sh)) & (DH - 1);
  const int Bb = vox >> (3 * sh);
  v8f acc = {};
  for (int p = 0; p < 4; ++p) {
    v16h a;
#pragma unroll
    for (int s = 0; s < 2; ++s) {
      const int t = 2 * p + s;
      const int dx = t >> 2, dy = (t >> 1) & 1, dz = t & 1;
      const unsigned nvx = (unsigned)(((((Bb << shi) | (2 * X + dx)) << shi) | (2 * Y + dy)) << shi | (2 * Z + dz));
      const v8h h = load_a8(in, nvx, hi);
#pragma unroll
      for (int q = 0; q < 8; ++q) a[s * 8 + q] = h[q];
    }
    const v8h b0 = *(const v8h*)(wlds + (size_t)((p * 2) * 16 + ln) * 16 + hi * 8);
    const v8h b1 = *(const v8h*)(wlds + (size_t)((p * 2 + 1) * 16 + ln) * 16 + hi * 8);
    const v16h b = __builtin_shufflevector(b0, b1, 0, 1, 2, 3, 4, 5, 6, 7, 8, 9, 10, 11, 12, 13, 14, 15);
    acc = __builtin_amdgcn_wmma_f32_16x16x32_f16(false, a, false, b, (short)0, acc, false, false);
  }
#pragma unroll
  for (int r = 0; r < 8; ++r) {
    const int ov = base + hi * 8 + r;
    const float mk = maskOut[ov] ? 1.0f : 0.0f;
    out[(size_t)ov * 16 + ln] = (_Float16)(acc[r] * mk);
  }
}

// Transposed conv k=2 s=2 VALID; blockIdx.y = output parity tap.
// One contributing tap per output-parity subgrid -> K upper half zero.
__global__ __launch_bounds__(128) void k_up_wmma(
    const _Float16* __restrict__ in, _Float16* __restrict__ out,
    const unsigned char* __restrict__ maskOut, const _Float16* __restrict__ wq, int sh) {
  const int tap = blockIdx.y;
  __shared__ __align__(64) _Float16 wlds[512];
  stage_weights(wq + (size_t)tap * 512, wlds, 1024);   // this tap's [2][16][16] slice
  const int lane = threadIdx.x & 31;
  const int tile = blockIdx.x * (blockDim.x >> 5) + (threadIdx.x >> 5);
  const int base = tile << 4;
  const int ln = lane & 15, hi = lane >> 4;
  v16h a = {};
  const v8h h = load_a8(in, (unsigned)(base + ln), hi);
#pragma unroll
  for (int q = 0; q < 8; ++q) a[q] = h[q];
  const v8h b0 = *(const v8h*)(wlds + (size_t)ln * 16 + hi * 8);
  const v8h b1 = *(const v8h*)(wlds + (size_t)(16 + ln) * 16 + hi * 8);
  const v16h b = __builtin_shufflevector(b0, b1, 0, 1, 2, 3, 4, 5, 6, 7, 8, 9, 10, 11, 12, 13, 14, 15);
  v8f acc = {};
  acc = __builtin_amdgcn_wmma_f32_16x16x32_f16(false, a, false, b, (short)0, acc, false, false);
  const int DH = 1 << sh, sho = sh + 1;
  const int px = tap >> 2, py = (tap >> 1) & 1, pz = tap & 1;
#pragma unroll
  for (int r = 0; r < 8; ++r) {
    const int iv = base + hi * 8 + r;
    const int Zh = iv & (DH - 1), Yh = (iv >> sh) & (DH - 1), Xh = (iv >> (2 * sh)) & (DH - 1);
    const int Bb = iv >> (3 * sh);
    const int ov = ((((Bb << sho) | (2 * Xh + px)) << sho) | (2 * Yh + py)) << sho | (2 * Zh + pz);
    const float mk = maskOut[ov] ? 1.0f : 0.0f;
    out[(size_t)ov * 16 + ln] = (_Float16)(acc[r] * mk);
  }
}

// ------------------------- epilogue ----------------------------------------
__global__ void k_hidden_copy(const _Float16* __restrict__ x, float* __restrict__ out) {
  int i = blockIdx.x * blockDim.x + threadIdx.x;
  if (i >= 4 * 8192) return;
  int b = i >> 13, rem = i & 8191, c = rem >> 9, s = rem & 511;
  out[i] = (float)x[(size_t)(b * 512 + s) * 16 + c];   // NCDHW flatten order
}

__global__ void k_out_conv(const _Float16* __restrict__ x, const unsigned char* __restrict__ mask,
                           const float* __restrict__ w, float* __restrict__ out, int nv) {
  int v = blockIdx.x * blockDim.x + threadIdx.x;
  if (v >= nv) return;
  float a = 0.0f;
  if (mask[v]) {
    const int Z = v & 63, Y = (v >> 6) & 63, X = (v >> 12) & 63, Bb = v >> 18;
    for (int t = 0; t < 27; ++t) {
      const int dx = t / 9 - 1, dy = (t / 3) % 3 - 1, dz = t % 3 - 1;
      const int nx = X + dx, ny = Y + dy, nz = Z + dz;
      if ((unsigned)nx < 64u && (unsigned)ny < 64u && (unsigned)nz < 64u) {
        const _Float16* p = x + (size_t)(((((Bb << 6) | nx) << 6) | ny) << 6 | nz) * 16;
#pragma unroll
        for (int c = 0; c < 16; ++c) a += (float)p[c] * w[t * 16 + c];
      }
    }
  }
  out[v] = a;
}

// ---------------------------------------------------------------------------
extern "C" void kernel_launch(void* const* d_in, const int* in_sizes, int n_in,
                              void* d_out, int out_size, void* d_ws, size_t ws_size,
                              hipStream_t stream) {
  const float* data      = (const float*)d_in[0];
  const int*   clust     = (const int*)  d_in[1];
  const float* w_prep    = (const float*)d_in[2];
  const float* enc_gamma = (const float*)d_in[3];
  const float* enc_beta  = (const float*)d_in[4];
  const float* enc_wsub  = (const float*)d_in[5];
  const float* enc_wdown = (const float*)d_in[6];
  const float* dec_gamma = (const float*)d_in[7];
  const float* dec_beta  = (const float*)d_in[8];
  const float* dec_wup   = (const float*)d_in[9];
  const float* dec_wsub3 = (const float*)d_in[10];
  const float* dec_wsub4 = (const float*)d_in[11];
  const float* wo        = (const float*)d_in[12];
  float* out = (float*)d_out;
  const int npts = in_sizes[1];

  const int NV0 = 4 * 64 * 64 * 64, NV1 = 4 * 32 * 32 * 32, NV2 = 4 * 16 * 16 * 16, NV3 = 4 * 8 * 8 * 8;

  char* p = (char*)d_ws;
  auto alloc = [&](size_t bytes) { char* r = p; p += (bytes + 255) & ~(size_t)255; return r; };

  float* cnt = (float*)alloc((size_t)NV0 * 4);
  unsigned char* mask0 = (unsigned char*)alloc(NV0);
  unsigned char* mask1 = (unsigned char*)alloc(NV1);
  unsigned char* mask2 = (unsigned char*)alloc(NV2);
  unsigned char* mask3 = (unsigned char*)alloc(NV3);
  _Float16* fA0 = (_Float16*)alloc((size_t)NV0 * 32);
  _Float16* fB0 = (_Float16*)alloc((size_t)NV0 * 32);
  _Float16* fA1 = (_Float16*)alloc((size_t)NV1 * 32);
  _Float16* fB1 = (_Float16*)alloc((size_t)NV1 * 32);
  _Float16* fA2 = (_Float16*)alloc((size_t)NV2 * 32);
  _Float16* fB2 = (_Float16*)alloc((size_t)NV2 * 32);
  _Float16* fA3 = (_Float16*)alloc((size_t)NV3 * 32);
  float* stats = (float*)alloc(256);
  float* ss    = (float*)alloc(256);
  _Float16 *wq_esub[3], *wq_edown[3], *wq_dup[3], *wq_dsub3[3], *wq_dsub4[3];
  for (int i = 0; i < 3; ++i) {
    wq_esub[i]  = (_Float16*)alloc(14 * 1024);
    wq_edown[i] = (_Float16*)alloc(4 * 1024);
    wq_dup[i]   = (_Float16*)alloc(8 * 1024);
    wq_dsub3[i] = (_Float16*)alloc(14 * 1024);
    wq_dsub4[i] = (_Float16*)alloc(32 * 1024);
  }

  for (int i = 0; i < 3; ++i) {
    k_wprep<<<(14 * 512 + 255) / 256, 256, 0, stream>>>(enc_wsub  + i * 27 * 256, wq_esub[i],  27, 0);
    k_wprep<<<(4  * 512 + 255) / 256, 256, 0, stream>>>(enc_wdown + i * 8  * 256, wq_edown[i],  8, 0);
    k_wprep<<<(8  * 512 + 255) / 256, 256, 0, stream>>>(dec_wup   + i * 8  * 256, wq_dup[i],    8, 1);
    k_wprep<<<(14 * 512 + 255) / 256, 256, 0, stream>>>(dec_wsub3 + i * 27 * 256, wq_dsub3[i], 27, 0);
    k_wprep<<<(32 * 512 + 255) / 256, 256, 0, stream>>>(dec_wsub4 + i * 64 * 256, wq_dsub4[i], 64, 0);
  }

  hipMemsetAsync(cnt, 0, (size_t)NV0 * 4, stream);
  k_scatter<<<(npts + 255) / 256, 256, 0, stream>>>(data, clust, cnt, npts);
  k_mask_from_cnt<<<(NV0 + 255) / 256, 256, 0, stream>>>(cnt, mask0, NV0);
  k_prep_conv<<<(NV0 + 255) / 256, 256, 0, stream>>>(cnt, mask0, w_prep, fA0, NV0);

  auto run_bn = [&](_Float16* x, unsigned char* m, int nv, const float* g, const float* b) {
    hipMemsetAsync(stats, 0, 33 * 4, stream);
    k_bn_stats<<<256, 256, 0, stream>>>(x, m, stats, nv);
    k_bn_finalize<<<1, 16, 0, stream>>>(stats, g, b, ss);
    k_bn_apply<<<(nv * 16 + 255) / 256, 256, 0, stream>>>(x, m, ss, nv);
  };

  _Float16* fa[4] = {fA0, fA1, fA2, fA3};
  _Float16* fb[3] = {fB0, fB1, fB2};
  unsigned char* mk[4] = {mask0, mask1, mask2, mask3};
  const int nvs[4] = {NV0, NV1, NV2, NV3};
  const int shs[4] = {6, 5, 4, 3};

  // encoder
  for (int i = 0; i < 3; ++i) {
    run_bn(fa[i], mk[i], nvs[i], enc_gamma + i * 16, enc_beta + i * 16);
    k_conv3_wmma<<<nvs[i] / 64, 128, 0, stream>>>(fa[i], fb[i], mk[i], wq_esub[i], shs[i]);
    k_mask_down<<<(nvs[i + 1] + 255) / 256, 256, 0, stream>>>(mk[i], mk[i + 1], shs[i + 1], nvs[i + 1]);
    k_down_wmma<<<nvs[i + 1] / 64, 128, 0, stream>>>(fb[i], fa[i + 1], mk[i + 1], wq_edown[i], shs[i + 1]);
  }

  k_hidden_copy<<<(4 * 8192 + 255) / 256, 256, 0, stream>>>(fA3, out);

  // decoder j=0 (level 3 -> 2)
  run_bn(fA3, mask3, NV3, dec_gamma + 0, dec_beta + 0);
  k_up_wmma<<<dim3(NV3 / 64, 8), 128, 0, stream>>>(fA3, fB2, mask2, wq_dup[0], 3);
  k_conv3_wmma<<<NV2 / 64, 128, 0, stream>>>(fB2, fA2, mask2, wq_dsub3[0], 4);
  k_sparsify<<<(NV2 + 255) / 256, 256, 0, stream>>>(fA2, mask2, NV2);
  k_conv4_wmma<<<NV2 / 64, 128, 0, stream>>>(fA2, fB2, mask2, wq_dsub4[0], 4);

  // decoder j=1 (level 2 -> 1)
  run_bn(fB2, mask2, NV2, dec_gamma + 16, dec_beta + 16);
  k_up_wmma<<<dim3(NV2 / 64, 8), 128, 0, stream>>>(fB2, fA1, mask1, wq_dup[1], 4);
  k_conv3_wmma<<<NV1 / 64, 128, 0, stream>>>(fA1, fB1, mask1, wq_dsub3[1], 5);
  k_sparsify<<<(NV1 + 255) / 256, 256, 0, stream>>>(fB1, mask1, NV1);
  k_conv4_wmma<<<NV1 / 64, 128, 0, stream>>>(fB1, fA1, mask1, wq_dsub4[1], 5);

  // decoder j=2 (level 1 -> 0)
  run_bn(fA1, mask1, NV1, dec_gamma + 32, dec_beta + 32);
  k_up_wmma<<<dim3(NV1 / 64, 8), 128, 0, stream>>>(fA1, fA0, mask0, wq_dup[2], 5);
  k_conv3_wmma<<<NV0 / 64, 128, 0, stream>>>(fA0, fB0, mask0, wq_dsub3[2], 6);
  k_sparsify<<<(NV0 + 255) / 256, 256, 0, stream>>>(fB0, mask0, NV0);
  k_conv4_wmma<<<NV0 / 64, 128, 0, stream>>>(fB0, fA0, mask0, wq_dsub4[2], 6);

  k_out_conv<<<(NV0 + 255) / 256, 256, 0, stream>>>(fA0, mask0, wo, out + 4 * 8192, NV0);
}
